// MultiQueryAttention_69114613727693
// MI455X (gfx1250) — matrix operand order repaired and checked by reading, not compiled, not run on hardware
//
#include <hip/hip_runtime.h>

#define DIM   2048
#define SEQ   2048
#define HEADS 16
#define HDIM  128

typedef __attribute__((ext_vector_type(16))) _Float16 v16h;
typedef __attribute__((ext_vector_type(8)))  _Float16 v8h;
typedef __attribute__((ext_vector_type(4)))  _Float16 v4h;
typedef __attribute__((ext_vector_type(8)))  float    v8f;

// ---------------------------------------------------------------------------
// WMMA wrapper: D = A(16x32 f16) * B(32x16 f16) + C(16x16 f32)
// ---------------------------------------------------------------------------
static __device__ __forceinline__ v8f wmma16(v16h a, v16h b, v8f c) {
  return __builtin_amdgcn_wmma_f32_16x16x32_f16(
      /*neg_a=*/false, a, /*neg_b=*/false, b,
      /*c_mod=*/(short)0, c, /*reuse_a=*/false, /*reuse_b=*/false);
}

// A-fragment (16x32, row-major source, leading dim ld in elements).
// Lane l: row = l&15; k = (l>>4)*16 + j
static __device__ __forceinline__ v16h load_a_frag(const _Float16* base, int ld) {
  const int lane = threadIdx.x & 31;
  const int row  = lane & 15;
  const int kg   = lane >> 4;
  const _Float16* p = base + (size_t)row * ld + kg * 16;
  v8h lo = *(const v8h*)p;          // 16B
  v8h hi = *(const v8h*)(p + 8);    // 16B
  v16h a;
#pragma unroll
  for (int j = 0; j < 8; ++j) { a[j] = lo[j]; a[8 + j] = hi[j]; }
  return a;
}

// B-fragment for C = A * W^T with W stored [N,K] row-major.
// B(k,n) = W[n0+n, k0+k]; lane l: n = l&15, k = (l>>4)*16 + j  (contiguous 32B)
static __device__ __forceinline__ v16h load_bt_frag(const _Float16* W, int ldw,
                                                    int n0, int k0) {
  const int lane = threadIdx.x & 31;
  const int col  = lane & 15;
  const int kg   = lane >> 4;
  const _Float16* p = W + (size_t)(n0 + col) * ldw + k0 + kg * 16;
  return *(const v16h*)p;           // one contiguous 32B load
}

// C/D store: lane l holds col = l&15, rows = (l>>4)*8 + r (r = VGPR index)
template <typename OUT_T>
static __device__ __forceinline__ void store_c(OUT_T* C, int ldc, int m0, int n0,
                                               v8f c) {
  const int lane = threadIdx.x & 31;
  const int col  = n0 + (lane & 15);
  const int rb   = m0 + (lane >> 4) * 8;
#pragma unroll
  for (int r = 0; r < 8; ++r) C[(size_t)(rb + r) * ldc + col] = (OUT_T)c[r];
}

// Transposed f16 store: Ct[n, m] = C[m, n]; per lane the 8 m's are contiguous.
static __device__ __forceinline__ void store_ct(_Float16* Ct, int ldm, int m0,
                                                int n0, v8f c) {
  const int lane = threadIdx.x & 31;
  const int col  = n0 + (lane & 15);
  const int rb   = m0 + (lane >> 4) * 8;
  _Float16* p = Ct + (size_t)col * ldm + rb;
#pragma unroll
  for (int r = 0; r < 8; ++r) p[r] = (_Float16)c[r];
}

// ---------------------------------------------------------------------------
// fp32 -> f16 conversion (vectorized x4)
// ---------------------------------------------------------------------------
__global__ void cvt_f32_to_f16(const float* __restrict__ in,
                               _Float16* __restrict__ out, int n) {
  int i = (blockIdx.x * blockDim.x + threadIdx.x) * 4;
  if (i < n) {
    float4 v = *(const float4*)(in + i);
    v4h h;
    h[0] = (_Float16)v.x; h[1] = (_Float16)v.y;
    h[2] = (_Float16)v.z; h[3] = (_Float16)v.w;
    *(v4h*)(out + i) = h;
  }
}

// ---------------------------------------------------------------------------
// GEMM (64x64 wave tile): C[M,N] = A[M,K] * W[N,K]^T. 16 WMMA per k-step,
// 8 fragment loads -> ~32 FLOP/byte from (L2-resident) memory.
// ---------------------------------------------------------------------------
template <typename OUT_T>
__global__ __launch_bounds__(256) void gemm_xwT_64(const _Float16* __restrict__ A,
                                                   const _Float16* __restrict__ W,
                                                   OUT_T* __restrict__ C,
                                                   int M, int N, int K) {
  const int wave = blockIdx.x * (blockDim.x >> 5) + (threadIdx.x >> 5);
  const int n64  = N >> 6;
  const int tm   = wave / n64;
  const int tn   = wave - tm * n64;
  if (tm * 64 >= M) return;          // wave-uniform; EXEC stays all-ones
  const int m0 = tm * 64, n0 = tn * 64;

  v8f acc[4][4] = {};
  for (int k0 = 0; k0 < K; k0 += 32) {
    v16h a[4], b[4];
#pragma unroll
    for (int i = 0; i < 4; ++i)
      a[i] = load_a_frag(A + (size_t)(m0 + 16 * i) * K + k0, K);
#pragma unroll
    for (int j = 0; j < 4; ++j)
      b[j] = load_bt_frag(W, K, n0 + 16 * j, k0);
#pragma unroll
    for (int i = 0; i < 4; ++i)
#pragma unroll
      for (int j = 0; j < 4; ++j)
        acc[i][j] = wmma16(a[i], b[j], acc[i][j]);
  }
#pragma unroll
  for (int i = 0; i < 4; ++i)
#pragma unroll
    for (int j = 0; j < 4; ++j)
      store_c(C, N, m0 + 16 * i, n0 + 16 * j, acc[i][j]);
}

// ---------------------------------------------------------------------------
// GEMM (32x32 wave tile) for the narrow K/V projections (N = 128).
// TRANS=true stores C^T (f16) so attention can load V fragments contiguously.
// ---------------------------------------------------------------------------
template <bool TRANS>
__global__ __launch_bounds__(256) void gemm_xwT_32(const _Float16* __restrict__ A,
                                                   const _Float16* __restrict__ W,
                                                   _Float16* __restrict__ C,
                                                   int M, int N, int K) {
  const int wave = blockIdx.x * (blockDim.x >> 5) + (threadIdx.x >> 5);
  const int n32  = N >> 5;
  const int tm   = wave / n32;
  const int tn   = wave - tm * n32;
  if (tm * 32 >= M) return;
  const int m0 = tm * 32, n0 = tn * 32;

  v8f c00 = {}, c01 = {}, c10 = {}, c11 = {};
  for (int k0 = 0; k0 < K; k0 += 32) {
    v16h a0 = load_a_frag(A + (size_t)m0 * K + k0, K);
    v16h a1 = load_a_frag(A + (size_t)(m0 + 16) * K + k0, K);
    v16h b0 = load_bt_frag(W, K, n0, k0);
    v16h b1 = load_bt_frag(W, K, n0 + 16, k0);
    c00 = wmma16(a0, b0, c00);
    c01 = wmma16(a0, b1, c01);
    c10 = wmma16(a1, b0, c10);
    c11 = wmma16(a1, b1, c11);
  }
  if (TRANS) {
    store_ct(C, M, m0,      n0,      c00);
    store_ct(C, M, m0,      n0 + 16, c01);
    store_ct(C, M, m0 + 16, n0,      c10);
    store_ct(C, M, m0 + 16, n0 + 16, c11);
  } else {
    store_c(C, N, m0,      n0,      c00);
    store_c(C, N, m0,      n0 + 16, c01);
    store_c(C, N, m0 + 16, n0,      c10);
    store_c(C, N, m0 + 16, n0 + 16, c11);
  }
}

// ---------------------------------------------------------------------------
// Flash-style MQA attention. One wave = (head h, 16 query rows).
// Q: [SEQ, DIM] f16 (head h -> cols h*128..+127); K: [SEQ, HDIM]; Vt: [HDIM, SEQ].
// 1/sqrt(128) folded into the Q fragments. Online softmax; P goes C-layout ->
// A-layout through a per-wave LDS tile.
// ---------------------------------------------------------------------------
#define LDS_STRIDE 40   // halfs; 80B rows keep 16B alignment for b128 LDS loads

__global__ __launch_bounds__(256) void mqa_attn(const _Float16* __restrict__ Q,
                                                const _Float16* __restrict__ Km,
                                                const _Float16* __restrict__ Vt,
                                                _Float16* __restrict__ O) {
  __shared__ __align__(16) _Float16 lds[8][16 * LDS_STRIDE];
  const int wlocal = threadIdx.x >> 5;
  const int gwave  = blockIdx.x * 8 + wlocal;
  const int h      = gwave >> 7;          // 128 query tiles per head
  const int qt     = gwave & 127;
  const int q0     = qt * 16;
  const int lane   = threadIdx.x & 31;

  // Q fragments, pre-scaled by 1/sqrt(HDIM)
  const _Float16 hscale = (_Float16)0.08838834764831845f;
  v16h aq[4];
#pragma unroll
  for (int t = 0; t < 4; ++t) {
    aq[t] = load_a_frag(Q + (size_t)q0 * DIM + h * HDIM + t * 32, DIM);
#pragma unroll
    for (int j = 0; j < 16; ++j) aq[t][j] *= hscale;
  }

  v8f acc[8] = {};                  // O accumulator: 16 x 128 f32
  float mrow[8], lrow[8];
#pragma unroll
  for (int r = 0; r < 8; ++r) { mrow[r] = -3.0e38f; lrow[r] = 0.0f; }

  _Float16* tile = lds[wlocal];

  for (int kb = 0; kb < SEQ; kb += 32) {
    // S = Q * K^T for 32 keys (two 16x16 tiles), contraction over hd = 128
    v8f s0 = {}, s1 = {};
#pragma unroll
    for (int t = 0; t < 4; ++t) {
      s0 = wmma16(aq[t], load_bt_frag(Km, HDIM, kb,      t * 32), s0);
      s1 = wmma16(aq[t], load_bt_frag(Km, HDIM, kb + 16, t * 32), s1);
    }

    float p0[8], p1[8], rmax[8];
#pragma unroll
    for (int r = 0; r < 8; ++r) {
      p0[r] = s0[r];
      p1[r] = s1[r];
      rmax[r] = fmaxf(p0[r], p1[r]);
    }
    // row-wise max across the 16 lanes holding one row's columns
#pragma unroll
    for (int r = 0; r < 8; ++r) {
      float v = rmax[r];
      v = fmaxf(v, __shfl_xor(v, 1));
      v = fmaxf(v, __shfl_xor(v, 2));
      v = fmaxf(v, __shfl_xor(v, 4));
      v = fmaxf(v, __shfl_xor(v, 8));
      rmax[r] = v;
    }
    float alpha[8];
#pragma unroll
    for (int r = 0; r < 8; ++r) {
      float mnew = fmaxf(mrow[r], rmax[r]);
      alpha[r]   = __expf(mrow[r] - mnew);
      mrow[r]    = mnew;
      p0[r] = __expf(p0[r] - mnew);
      p1[r] = __expf(p1[r] - mnew);
      float s = p0[r] + p1[r];
      s += __shfl_xor(s, 1);
      s += __shfl_xor(s, 2);
      s += __shfl_xor(s, 4);
      s += __shfl_xor(s, 8);
      lrow[r] = lrow[r] * alpha[r] + s;
    }
#pragma unroll
    for (int t = 0; t < 8; ++t)
#pragma unroll
      for (int r = 0; r < 8; ++r) acc[t][r] *= alpha[r];

    // P (C-layout) -> LDS [16 rows x 32 keys]
    {
      const int c0 = lane & 15;
      const int rb = (lane >> 4) * 8;
#pragma unroll
      for (int r = 0; r < 8; ++r) {
        tile[(rb + r) * LDS_STRIDE + c0]      = (_Float16)p0[r];
        tile[(rb + r) * LDS_STRIDE + 16 + c0] = (_Float16)p1[r];
      }
    }
    __syncthreads();                       // uniform trip count across waves
    v16h ap = load_a_frag(tile, LDS_STRIDE);   // re-read as A-fragment
    __syncthreads();

    // O += P * V; V fragments from Vt are contiguous 32B loads per lane
#pragma unroll
    for (int t = 0; t < 8; ++t)
      acc[t] = wmma16(ap, load_bt_frag(Vt, SEQ, t * 16, kb), acc[t]);
  }

  // normalize by softmax denominator, store f16 in [seq, dim] layout
  float inv[8];
#pragma unroll
  for (int r = 0; r < 8; ++r) inv[r] = 1.0f / lrow[r];
  const int col = lane & 15;
  const int rb  = (lane >> 4) * 8;
#pragma unroll
  for (int t = 0; t < 8; ++t)
#pragma unroll
    for (int r = 0; r < 8; ++r)
      O[(size_t)(q0 + rb + r) * DIM + h * HDIM + t * 16 + col] =
          (_Float16)(acc[t][r] * inv[r]);
}

// ---------------------------------------------------------------------------
// Host-side orchestration
// ---------------------------------------------------------------------------
extern "C" void kernel_launch(void* const* d_in, const int* in_sizes, int n_in,
                              void* d_out, int out_size, void* d_ws, size_t ws_size,
                              hipStream_t stream) {
  const float* x  = (const float*)d_in[0];
  const float* Wq = (const float*)d_in[1];
  const float* Wk = (const float*)d_in[2];
  const float* Wv = (const float*)d_in[3];
  const float* Wo = (const float*)d_in[4];
  float* out = (float*)d_out;

  // Workspace carve-out (~42 MB total, f16)
  char* ws = (char*)d_ws;
  size_t off = 0;
  auto carve = [&](size_t bytes) -> _Float16* {
    char* p = ws + off;
    off += (bytes + 255) & ~(size_t)255;
    return (_Float16*)p;
  };
  _Float16* xh  = carve((size_t)SEQ * DIM * 2);
  _Float16* wqh = carve((size_t)DIM * DIM * 2);
  _Float16* woh = carve((size_t)DIM * DIM * 2);
  _Float16* wkh = carve((size_t)HDIM * DIM * 2);
  _Float16* wvh = carve((size_t)HDIM * DIM * 2);
  _Float16* Qh  = carve((size_t)SEQ * DIM * 2);
  _Float16* Kh  = carve((size_t)SEQ * HDIM * 2);   // [SEQ, HDIM]
  _Float16* Vth = carve((size_t)HDIM * SEQ * 2);   // [HDIM, SEQ] (transposed)
  _Float16* Ah  = carve((size_t)SEQ * DIM * 2);
  (void)ws_size; (void)in_sizes; (void)n_in; (void)out_size;

  auto cvt = [&](const float* src, _Float16* dst, int n) {
    cvt_f32_to_f16<<<(n / 4 + 255) / 256, 256, 0, stream>>>(src, dst, n);
  };
  cvt(x,  xh,  SEQ * DIM);
  cvt(Wq, wqh, DIM * DIM);
  cvt(Wk, wkh, HDIM * DIM);
  cvt(Wv, wvh, HDIM * DIM);
  cvt(Wo, woh, DIM * DIM);

  // Q projection (2048^3): 64x64 wave tiles -> 1024 waves
  {
    int waves = (SEQ / 64) * (DIM / 64);
    gemm_xwT_64<_Float16><<<waves / 8, 256, 0, stream>>>(xh, wqh, Qh, SEQ, DIM, DIM);
  }
  // K, V projections (N = 128): 32x32 wave tiles; V stored transposed
  {
    int waves = (SEQ / 32) * (HDIM / 32);           // 256
    gemm_xwT_32<false><<<waves / 8, 256, 0, stream>>>(xh, wkh, Kh,  SEQ, HDIM, DIM);
    gemm_xwT_32<true ><<<waves / 8, 256, 0, stream>>>(xh, wvh, Vth, SEQ, HDIM, DIM);
  }

  // MQA flash attention: 16 heads x 128 query tiles = 2048 waves
  mqa_attn<<<(HEADS * (SEQ / 16)) / 8, 256, 0, stream>>>(Qh, Kh, Vth, Ah);

  // Output projection (2048^3), fp32 result
  {
    int waves = (SEQ / 64) * (DIM / 64);
    gemm_xwT_64<float><<<waves / 8, 256, 0, stream>>>(Ah, woh, out, SEQ, DIM, DIM);
  }
}
